// MultiHeadAttention_76459007803717
// MI455X (gfx1250) — compile-verified, matrix operand order checked by reading, not run on hardware
//
#include <hip/hip_runtime.h>
#include <hip/hip_bf16.h>

// ---------------------------------------------------------------------------
// MHA forward for B=4, S=1024, H=1024, NH=16, HD=64 on gfx1250 (wave32, WMMA)
// GEMMs: V_WMMA_F32_16X16X32_BF16, 4-slab async global->LDS pipeline
// (GLOBAL_LOAD_ASYNC_TO_LDS_B128, s_wait_asynccnt 2 keeps 2 slabs in flight).
// ---------------------------------------------------------------------------

typedef __attribute__((ext_vector_type(16))) __bf16 v16bf;
typedef __attribute__((ext_vector_type(8)))  float  v8f;
typedef __attribute__((address_space(3))) unsigned short lds_us;

#define B_SZ 4
#define S_SZ 1024
#define H_SZ 1024
#define NH_SZ 16
#define HD_SZ 64
#define M_SZ (B_SZ * S_SZ)      // 4096

__device__ __forceinline__ unsigned short f2bf(float f) {
  union { float f; unsigned u; } v; v.f = f;
  unsigned u = v.u;
  u += 0x7FFFu + ((u >> 16) & 1u);     // round to nearest even
  return (unsigned short)(u >> 16);
}

union FragU { uint4 u[2]; v16bf v; };

// Load a 16x32 bf16 A-fragment (or the mirror B-fragment) from a row-major
// matrix. ISA layout: lane l holds row (l&15); K chunks [k0+hi*8, +8) and
// [k0+16+hi*8, +8) where hi = l>>4. Two 16B loads per lane.
__device__ __forceinline__ v16bf load_frag(const unsigned short* __restrict__ base,
                                           int row0, int ld, int k0, int lane) {
  int l = lane & 15, hi = lane >> 4;
  const unsigned short* p = base + (size_t)(row0 + l) * ld + k0 + hi * 8;
  FragU t;
  t.u[0] = *(const uint4*)(p);
  t.u[1] = *(const uint4*)(p + 16);
  return t.v;
}

__device__ __forceinline__ v8f wmma_bf16(v16bf a, v16bf b, v8f c) {
  return __builtin_amdgcn_wmma_f32_16x16x32_bf16(false, a, false, b,
                                                 (short)0, c, false, false);
}

// Async global -> LDS copy, 16 bytes per lane, tracked by ASYNCcnt.
__device__ __forceinline__ void async_copy_b128(lds_us* lds_dst,
                                                const unsigned short* gsrc) {
  asm volatile("global_load_async_to_lds_b128 %0, %1, off"
               :: "v"(lds_dst), "v"(gsrc)
               : "memory");
}

__device__ __forceinline__ void wait_async0() {
  asm volatile("s_wait_asynccnt 0" ::: "memory");
}
// ASYNCcnt retires in issue order: <=2 retires the oldest slab while the
// most recently issued slab (2 copies) stays in flight.
__device__ __forceinline__ void wait_async2() {
  asm volatile("s_wait_asynccnt 2" ::: "memory");
}

// ---------------------------------------------------------------------------
// fp32 -> bf16 conversion (4 elems/thread, vectorized)
// ---------------------------------------------------------------------------
__global__ void k_f32_to_bf16(const float* __restrict__ src,
                              unsigned short* __restrict__ dst, int n) {
  int i = (blockIdx.x * blockDim.x + threadIdx.x) * 4;
  if (i + 3 < n) {
    float4 f = *(const float4*)(src + i);
    union { unsigned short s[4]; uint2 u; } o;
    o.s[0] = f2bf(f.x); o.s[1] = f2bf(f.y);
    o.s[2] = f2bf(f.z); o.s[3] = f2bf(f.w);
    *(uint2*)(dst + i) = o.u;
  }
}

// ---------------------------------------------------------------------------
// Unified projection GEMM: C = A * W^T + bias, M=4096, N=1024, K=1024.
// mode 0/1/2: A = Xb, epilogue -> Qb / Kb ([B,NH,S,HD] bf16) / Vtb
//             ([B,NH,HD,S] bf16, transposed for the P*V B-fragments).
// mode 3:     A = Ctxb, epilogue -> fp32 out + bo.
// Block = 64x64 C tile, 8 waves, 2 C-tiles/wave. A and B 64x32 k-slabs flow
// through a 4-slot LDS ring filled by per-lane GLOBAL_LOAD_ASYNC_TO_LDS_B128;
// s_wait_asynccnt 2 keeps two slabs in flight, so each slab has two full
// iterations of WMMA work between issue and the wait that consumes it.
// ---------------------------------------------------------------------------
__global__ __launch_bounds__(256) void k_gemm(
    int mode_base,
    const unsigned short* __restrict__ Xb,
    const unsigned short* __restrict__ Ctxb,
    const unsigned short* __restrict__ Wqb,
    const unsigned short* __restrict__ Wkb,
    const unsigned short* __restrict__ Wvb,
    const unsigned short* __restrict__ Wob,
    const float* __restrict__ bq, const float* __restrict__ bk,
    const float* __restrict__ bv, const float* __restrict__ bo,
    unsigned short* __restrict__ Qb, unsigned short* __restrict__ Kb,
    unsigned short* __restrict__ Vtb, float* __restrict__ out) {
  __shared__ __align__(16) unsigned short Asm[4][64 * 32];   // 4 x 4KB
  __shared__ __align__(16) unsigned short Bsm[4][64 * 32];   // 4 x 4KB

  int mode = mode_base + blockIdx.z;
  const unsigned short* A = (mode < 3) ? Xb : Ctxb;
  const unsigned short* W =
      (mode == 0) ? Wqb : (mode == 1) ? Wkb : (mode == 2) ? Wvb : Wob;
  const float* bias =
      (mode == 0) ? bq : (mode == 1) ? bk : (mode == 2) ? bv : bo;

  int m0 = blockIdx.x * 64, n0 = blockIdx.y * 64;
  int tid = threadIdx.x;
  int w = tid >> 5, lane = tid & 31;
  int mt = w & 3, ntA = (w >> 2) * 2, ntB = ntA + 1;
  int rowA = m0 + mt * 16;

  // staging assignment: thread t moves 8 elems of A row (t>>2) and B row (t>>2)
  int srow = tid >> 2, sseg = (tid & 3) * 8;
  const unsigned short* gA = A + (size_t)(m0 + srow) * H_SZ + sseg;
  const unsigned short* gB = W + (size_t)(n0 + srow) * H_SZ + sseg;
  int soff = srow * 32 + sseg;

  const int KT = H_SZ / 32;   // 32 k-slabs

  // prologue: stage slabs 0 and 1 (2 async copy pairs in flight)
  async_copy_b128((lds_us*)&Asm[0][soff], gA);
  async_copy_b128((lds_us*)&Bsm[0][soff], gB);
  async_copy_b128((lds_us*)&Asm[1][soff], gA + 32);
  async_copy_b128((lds_us*)&Bsm[1][soff], gB + 32);

  v8f c0 = {}, c1 = {};
  for (int kt = 0; kt < KT; ++kt) {
    int cur = kt & 3;
    // slab kt resident (issued 2 iters ago); slab kt+1 may stay in flight
    if (kt < KT - 1) wait_async2(); else wait_async0();
    __syncthreads();
    if (kt + 2 < KT) {      // refill ring slot read two iterations ago
      int k = (kt + 2) * 32;
      async_copy_b128((lds_us*)&Asm[(kt + 2) & 3][soff], gA + k);
      async_copy_b128((lds_us*)&Bsm[(kt + 2) & 3][soff], gB + k);
    }
    v16bf a  = load_frag(&Asm[cur][0], mt * 16, 32, 0, lane);
    v16bf b0 = load_frag(&Bsm[cur][0], ntA * 16, 32, 0, lane);
    v16bf b1 = load_frag(&Bsm[cur][0], ntB * 16, 32, 0, lane);
    c0 = wmma_bf16(a, b0, c0);
    c1 = wmma_bf16(a, b1, c1);
    // slot `cur` is next overwritten by copies issued at iteration kt+2,
    // i.e. after two more wait+barrier points: all waves' ds_loads of this
    // slab are long since consumed into WMMA source registers.
  }

  int l = lane & 15, hi = lane >> 4;
  for (int half = 0; half < 2; ++half) {
    v8f c = half ? c1 : c0;
    int nt = half ? ntB : ntA;
    int ncol = n0 + nt * 16 + l;
    float bval = bias[ncol];
    int h = ncol >> 6, hd = ncol & 63;
    for (int r = 0; r < 8; ++r) {
      int m = rowA + hi * 8 + r;
      if (mode == 3) {
        out[(size_t)m * H_SZ + ncol] = c[r] + bval;
      } else {
        int b = m >> 10, s = m & 1023;
        unsigned short bf = f2bf(c[r] + bval);
        if (mode == 2) {
          Vtb[((size_t)((b * NH_SZ + h) * HD_SZ + hd)) * S_SZ + s] = bf;
        } else {
          unsigned short* dst = (mode == 0) ? Qb : Kb;
          dst[((size_t)((b * NH_SZ + h) * S_SZ + s)) * HD_SZ + hd] = bf;
        }
      }
    }
  }
}

// ---------------------------------------------------------------------------
// Attention: one block per (b, h, 16-row q tile). 8 waves.
// Phase 1: scores 16x1024 in registers (8 tiles/wave), scale + mask.
// Phase 2: row softmax (shfl_xor within 16-lane groups + LDS cross-wave).
// Phase 3: write probs fp32 to HBM once, stage bf16 P in LDS.
// Phase 4: ctx = P * V from LDS (A) and transposed V (B); K split over waves.
// ---------------------------------------------------------------------------
__global__ __launch_bounds__(256) void k_attn(
    const unsigned short* __restrict__ Qb,
    const unsigned short* __restrict__ Kb,
    const unsigned short* __restrict__ Vtb,
    const float* __restrict__ mask,
    float* __restrict__ probs,
    unsigned short* __restrict__ Ctxb) {
  __shared__ __align__(16) unsigned short Pbf[16 * S_SZ];  // 32 KB
  __shared__ float redmx[8][16];
  __shared__ float redsm[8][16];
  __shared__ float cred[4][32][8];                          // 4 KB

  int bid = blockIdx.x;
  int qt = bid & 63;
  int bh = bid >> 6;                // b*16 + h
  int b = bh >> 4, h = bh & 15;

  int w = threadIdx.x >> 5, lane = threadIdx.x & 31;
  int l = lane & 15, hi = lane >> 4;

  const unsigned short* Qbase = Qb + (size_t)bh * S_SZ * HD_SZ;
  const unsigned short* Kbase = Kb + (size_t)bh * S_SZ * HD_SZ;
  const unsigned short* Vbase = Vtb + (size_t)bh * HD_SZ * S_SZ;

  // Q fragments for this 16-row tile (K-dim = HD = 64 -> two 32-chunks)
  v16bf aq0 = load_frag(Qbase, qt * 16, HD_SZ, 0, lane);
  v16bf aq1 = load_frag(Qbase, qt * 16, HD_SZ, 32, lane);

  // ---- scores: wave w owns column tiles {t*8 + w} ----
  float sc[8][8];
  for (int t = 0; t < 8; ++t) {
    int ct = t * 8 + w;
    v16bf bk0 = load_frag(Kbase, ct * 16, HD_SZ, 0, lane);
    v16bf bk1 = load_frag(Kbase, ct * 16, HD_SZ, 32, lane);
    v8f c = {};
    c = wmma_bf16(aq0, bk0, c);
    c = wmma_bf16(aq1, bk1, c);
    float mv = mask[b * S_SZ + ct * 16 + l];
    float term = (1.0f - mv) * -10000.0f;
    for (int r = 0; r < 8; ++r) sc[t][r] = c[r] * 0.125f + term;
  }

  // ---- row max ----
  float rmax[8];
  for (int r = 0; r < 8; ++r) {
    float v = sc[0][r];
    for (int t = 1; t < 8; ++t) v = fmaxf(v, sc[t][r]);
    for (int off = 1; off < 16; off <<= 1) v = fmaxf(v, __shfl_xor(v, off, 32));
    rmax[r] = v;
  }
  if (lane == 0)  for (int r = 0; r < 8; ++r) redmx[w][r] = rmax[r];
  if (lane == 16) for (int r = 0; r < 8; ++r) redmx[w][8 + r] = rmax[r];
  __syncthreads();
  float gmax[8];
  for (int r = 0; r < 8; ++r) {
    float v = redmx[0][hi * 8 + r];
    for (int ww = 1; ww < 8; ++ww) v = fmaxf(v, redmx[ww][hi * 8 + r]);
    gmax[r] = v;
  }

  // ---- exp + row sum ----
  float rsum[8];
  for (int r = 0; r < 8; ++r) rsum[r] = 0.0f;
  for (int t = 0; t < 8; ++t)
    for (int r = 0; r < 8; ++r) {
      float e = __expf(sc[t][r] - gmax[r]);
      sc[t][r] = e;
      rsum[r] += e;
    }
  for (int r = 0; r < 8; ++r) {
    float v = rsum[r];
    for (int off = 1; off < 16; off <<= 1) v += __shfl_xor(v, off, 32);
    rsum[r] = v;
  }
  if (lane == 0)  for (int r = 0; r < 8; ++r) redsm[w][r] = rsum[r];
  if (lane == 16) for (int r = 0; r < 8; ++r) redsm[w][8 + r] = rsum[r];
  __syncthreads();
  float ginv[8];
  for (int r = 0; r < 8; ++r) {
    float v = 0.0f;
    for (int ww = 0; ww < 8; ++ww) v += redsm[ww][hi * 8 + r];
    ginv[r] = 1.0f / v;
  }

  // ---- normalize, write probs once, stage bf16 P in LDS ----
  float* prow = probs + (size_t)bh * S_SZ * S_SZ + (size_t)qt * 16 * S_SZ;
  for (int t = 0; t < 8; ++t) {
    int ct = t * 8 + w;
    int col = ct * 16 + l;
    for (int r = 0; r < 8; ++r) {
      int row = hi * 8 + r;
      float p = sc[t][r] * ginv[r];
      prow[(size_t)row * S_SZ + col] = p;
      Pbf[row * S_SZ + col] = f2bf(p);
    }
  }
  __syncthreads();

  // ---- ctx = P (16x1024, LDS) * V (1024x64, via Vt) ----
  int nt = w & 3;          // which 16-col slice of HD
  int kh = w >> 2;         // K split: waves 0-3 low half, 4-7 high half
  v8f acc = {};
  for (int kc = kh * 512; kc < kh * 512 + 512; kc += 32) {
    v16bf a  = load_frag(Pbf, 0, S_SZ, kc, lane);
    v16bf bv = load_frag(Vbase, nt * 16, S_SZ, kc, lane);
    acc = wmma_bf16(a, bv, acc);
  }
  __syncthreads();
  if (kh == 1)
    for (int r = 0; r < 8; ++r) cred[nt][lane][r] = acc[r];
  __syncthreads();
  if (kh == 0) {
    for (int r = 0; r < 8; ++r) acc[r] += cred[nt][lane][r];
    for (int r = 0; r < 8; ++r) {
      int srow = qt * 16 + hi * 8 + r;
      int ncol = h * HD_SZ + nt * 16 + l;
      Ctxb[((size_t)(b * S_SZ + srow)) * H_SZ + ncol] = f2bf(acc[r]);
    }
  }
}

// ---------------------------------------------------------------------------
extern "C" void kernel_launch(void* const* d_in, const int* in_sizes, int n_in,
                              void* d_out, int out_size, void* d_ws, size_t ws_size,
                              hipStream_t stream) {
  const float* hs   = (const float*)d_in[0];
  const float* mask = (const float*)d_in[1];
  const float* Wq   = (const float*)d_in[2];
  const float* bq   = (const float*)d_in[3];
  const float* Wk   = (const float*)d_in[4];
  const float* bk   = (const float*)d_in[5];
  const float* Wv   = (const float*)d_in[6];
  const float* bv   = (const float*)d_in[7];
  const float* Wo   = (const float*)d_in[8];
  const float* bo   = (const float*)d_in[9];

  float* out   = (float*)d_out;
  float* probs = out + (size_t)M_SZ * H_SZ;   // (out, probs) concatenated

  // workspace layout (bf16 buffers), total 48 MB
  unsigned short* Xb   = (unsigned short*)d_ws;          // 4096x1024
  unsigned short* Wqb  = Xb   + (size_t)M_SZ * H_SZ;     // 1024x1024
  unsigned short* Wkb  = Wqb  + (size_t)H_SZ * H_SZ;
  unsigned short* Wvb  = Wkb  + (size_t)H_SZ * H_SZ;
  unsigned short* Wob  = Wvb  + (size_t)H_SZ * H_SZ;
  unsigned short* Qb   = Wob  + (size_t)H_SZ * H_SZ;     // [B,NH,S,HD]
  unsigned short* Kb   = Qb   + (size_t)4194304;         // B*NH*S*HD
  unsigned short* Vtb  = Kb   + (size_t)4194304;         // [B,NH,HD,S]
  unsigned short* Ctxb = Vtb  + (size_t)4194304;         // [B,S,H]

  // fp32 -> bf16 staging
  k_f32_to_bf16<<<(M_SZ * H_SZ) / 1024, 256, 0, stream>>>(hs, Xb, M_SZ * H_SZ);
  k_f32_to_bf16<<<(H_SZ * H_SZ) / 1024, 256, 0, stream>>>(Wq, Wqb, H_SZ * H_SZ);
  k_f32_to_bf16<<<(H_SZ * H_SZ) / 1024, 256, 0, stream>>>(Wk, Wkb, H_SZ * H_SZ);
  k_f32_to_bf16<<<(H_SZ * H_SZ) / 1024, 256, 0, stream>>>(Wv, Wvb, H_SZ * H_SZ);
  k_f32_to_bf16<<<(H_SZ * H_SZ) / 1024, 256, 0, stream>>>(Wo, Wob, H_SZ * H_SZ);

  // QKV projections (modes 0,1,2 via blockIdx.z)
  k_gemm<<<dim3(M_SZ / 64, H_SZ / 64, 3), 256, 0, stream>>>(
      0, Xb, Ctxb, Wqb, Wkb, Wvb, Wob, bq, bk, bv, bo, Qb, Kb, Vtb, out);

  // fused scores + softmax + probs + ctx
  k_attn<<<B_SZ * NH_SZ * (S_SZ / 16), 256, 0, stream>>>(
      Qb, Kb, Vtb, mask, probs, Ctxb);

  // output projection (mode 3)
  k_gemm<<<dim3(M_SZ / 64, H_SZ / 64, 1), 256, 0, stream>>>(
      3, Xb, Ctxb, Wqb, Wkb, Wvb, Wob, bq, bk, bv, bo, Qb, Kb, Vtb, out);
}
